// GATModule_58342835749557
// MI455X (gfx1250) — matrix operand order repaired and checked by reading, not compiled, not run on hardware
//
#include <hip/hip_runtime.h>

typedef __attribute__((ext_vector_type(16))) _Float16 v16h;
typedef __attribute__((ext_vector_type(8)))  _Float16 v8h;
typedef __attribute__((ext_vector_type(8)))  float    v8f;

#define N_NODES 100000
#define N_EDGES 1600000
#define DIM 128
#define NH 8
#define ROWS 64           // rows per GEMM block (4 row-tiles of 16)

// ---------------------------------------------------------------------------
// Order-preserving float <-> uint mapping for atomicMax on floats
// ---------------------------------------------------------------------------
__device__ __forceinline__ unsigned fmap(float x) {
    unsigned u = __float_as_uint(x);
    return (u & 0x80000000u) ? ~u : (u | 0x80000000u);
}
__device__ __forceinline__ float funmap(unsigned u) {
    unsigned b = (u & 0x80000000u) ? (u ^ 0x80000000u) : ~u;
    return __uint_as_float(b);
}

// ---------------------------------------------------------------------------
// Per-wave GEMM: 64x16 output strip = A[64,128] (row-major f16 LDS) times
// B^T column tile (LDS bt[n][k]); v_wmma_f32_16x16x32_f16, accumulators for
// 4 row-tiles, B fragment loaded once per K-step and reused 4x.
// Fragment layouts used here (contiguous k in A and B, not the ISA table order):
//   A  : lanes 0-15 row=lane, K 0..15 contiguous;
//        lanes 16-31 same rows, K 16..31.
//   B  : lanes 0-15 col=lane, K 0..15 contiguous; lanes 16-31 K 16..31.
//   C/D: col = lane&15, rows = i + 8*(lane>=16).
// ---------------------------------------------------------------------------
__device__ __forceinline__ void wmma_strip_64x16(const _Float16* lds_a,
                                                 const _Float16* lds_bt,
                                                 int col_tile, int lane,
                                                 v8f c[4])
{
    const int half_sel = (lane >> 4) & 1;
    const int r        = lane & 15;
    const int a_ko     = half_sel * 16;
    const int b_ko     = half_sel * 16;
    const _Float16* brow = lds_bt + (col_tile * 16 + r) * DIM;
#pragma unroll
    for (int kk = 0; kk < DIM; kk += 32) {
        v16h b = *(const v16h*)(brow + kk + b_ko);
#pragma unroll
        for (int rt = 0; rt < 4; ++rt) {
            const _Float16* arow = lds_a + (rt * 16 + r) * DIM;
            v8h alo = *(const v8h*)(arow + kk + a_ko);
            v8h ahi = *(const v8h*)(arow + kk + 8 + a_ko);
            v16h a  = __builtin_shufflevector(alo, ahi,
                        0,1,2,3,4,5,6,7,8,9,10,11,12,13,14,15);
            c[rt] = __builtin_amdgcn_wmma_f32_16x16x32_f16(
                        false, a, false, b, (short)0, c[rt], false, false);
        }
    }
}

// ---------------------------------------------------------------------------
// Kernel 0: per-launch init. Transpose+convert the 3 square weights to f16,
// zero denom/agg, reset smax. 50000 blocks x 256 = N*DIM threads.
// ---------------------------------------------------------------------------
__global__ __launch_bounds__(256)
void k_init(const float* __restrict__ w_in, const float* __restrict__ w1,
            const float* __restrict__ w2,
            _Float16* __restrict__ w_in_t, _Float16* __restrict__ w1_t,
            _Float16* __restrict__ w2_t,
            unsigned* __restrict__ smax_u, float* __restrict__ denom,
            float* __restrict__ agg)
{
    int idx = blockIdx.x * 256 + threadIdx.x;
    if (idx < DIM * DIM) {
        int k = idx >> 7, c = idx & 127;      // w[k][c] row-major
        w_in_t[c * DIM + k] = (_Float16)w_in[idx];
        w1_t [c * DIM + k] = (_Float16)w1[idx];
        w2_t [c * DIM + k] = (_Float16)w2[idx];
    }
    if (idx < N_NODES * NH) { smax_u[idx] = 0u; denom[idx] = 0.0f; }
    if (idx < N_NODES * DIM) agg[idx] = 0.0f;
}

// ---------------------------------------------------------------------------
// Kernel 1: h = x @ w_in + b_in  (WMMA), fused a_u = h@w_au + b_au,
// a_v = h@w_av. One block = 64 rows; 8 waves = 8 column tiles x 4 row tiles.
// ---------------------------------------------------------------------------
__global__ __launch_bounds__(256)
void k_node_linear(const float* __restrict__ x, const float* __restrict__ b_in,
                   const float* __restrict__ w_au, const float* __restrict__ b_au,
                   const float* __restrict__ w_av,
                   const _Float16* __restrict__ w_in_t,
                   float* __restrict__ h, float* __restrict__ a_u,
                   float* __restrict__ a_v)
{
    __shared__ __align__(32) _Float16 s_x[ROWS * DIM];     // 16 KB
    __shared__ __align__(32) _Float16 s_bt[DIM * DIM];     // 32 KB
    __shared__ __align__(32) float    s_h[ROWS * DIM];     // 32 KB

    const int tid  = threadIdx.x;
    const int row0 = blockIdx.x * ROWS;

    for (int i = tid; i < ROWS * DIM; i += 256) {
        int r = i >> 7;
        s_x[i] = (row0 + r < N_NODES)
                   ? (_Float16)x[(size_t)(row0 + r) * DIM + (i & 127)]
                   : (_Float16)0.0f;
    }
    { // stage transposed f16 weight (32 KB) into LDS
        const uint4* wsrc = (const uint4*)w_in_t;
        uint4*       wdst = (uint4*)s_bt;
        for (int i = tid; i < (DIM * DIM) / 8; i += 256) wdst[i] = wsrc[i];
    }
    __syncthreads();

    const int wave = tid >> 5, lane = tid & 31;
    v8f c[4] = {};
    wmma_strip_64x16(s_x, s_bt, wave, lane, c);

    const int n     = lane & 15;
    const int mbase = (lane >> 4) * 8;
    const int col   = wave * 16 + n;
    const float bias = b_in[col];
#pragma unroll
    for (int rt = 0; rt < 4; ++rt)
#pragma unroll
        for (int i = 0; i < 8; ++i)
            s_h[(rt * 16 + mbase + i) * DIM + col] = c[rt][i] + bias;
    __syncthreads();

    // cooperative coalesced store of h (float4 per lane)
    for (int i = tid; i < ROWS * (DIM / 4); i += 256) {
        int r = i >> 5, c4 = (i & 31) * 4;
        if (row0 + r < N_NODES)
            *(float4*)&h[(size_t)(row0 + r) * DIM + c4] =
                *(const float4*)&s_h[r * DIM + c4];
    }

    // skinny projections: 64 rows x 8 heads x {u,v} = 1024 dots, 4 per thread
    {
        const int q    = tid & 15;
        const int hh   = q & 7;
        const bool isv = (q >> 3) != 0;
        const float* w = isv ? w_av : w_au;
#pragma unroll
        for (int j = 0; j < 4; ++j) {
            int r = (tid >> 4) + 16 * j;
            if (row0 + r >= N_NODES) continue;
            float acc = 0.0f;
#pragma unroll 8
            for (int k = 0; k < DIM; ++k)
                acc += s_h[r * DIM + k] * w[k * NH + hh];
            if (isv) a_v[(size_t)(row0 + r) * NH + hh] = acc;
            else     a_u[(size_t)(row0 + r) * NH + hh] = acc + b_au[hh];
        }
    }
}

// ---------------------------------------------------------------------------
// Kernel 2: per-(edge,head) score = leaky_relu(a_u[src]+a_v[dst]); segment max
// ---------------------------------------------------------------------------
__global__ __launch_bounds__(256)
void k_edge_max(const int* __restrict__ src, const int* __restrict__ dst,
                const float* __restrict__ a_u, const float* __restrict__ a_v,
                unsigned* __restrict__ smax_u)
{
    int idx = blockIdx.x * 256 + threadIdx.x;
    if (idx >= N_EDGES * NH) return;
    int e = idx >> 3, hh = idx & 7;
    int s = src[e], d = dst[e];
    float sc = a_u[(size_t)s * NH + hh] + a_v[(size_t)d * NH + hh];
    sc = sc > 0.0f ? sc : 0.2f * sc;
    atomicMax(&smax_u[(size_t)d * NH + hh], fmap(sc));
}

// ---------------------------------------------------------------------------
// Kernel 3: ex = exp(score - smax[dst]); store; denom[dst] += ex
// ---------------------------------------------------------------------------
__global__ __launch_bounds__(256)
void k_edge_exp(const int* __restrict__ src, const int* __restrict__ dst,
                const float* __restrict__ a_u, const float* __restrict__ a_v,
                const unsigned* __restrict__ smax_u,
                float* __restrict__ ex, float* __restrict__ denom)
{
    int idx = blockIdx.x * 256 + threadIdx.x;
    if (idx >= N_EDGES * NH) return;
    int e = idx >> 3, hh = idx & 7;
    int s = src[e], d = dst[e];
    float sc = a_u[(size_t)s * NH + hh] + a_v[(size_t)d * NH + hh];
    sc = sc > 0.0f ? sc : 0.2f * sc;
    float mx = funmap(smax_u[(size_t)d * NH + hh]);
    float v  = __expf(sc - mx);
    ex[(size_t)e * NH + hh] = v;
    atomicAdd(&denom[(size_t)d * NH + hh], v);
}

// ---------------------------------------------------------------------------
// Kernel 4: agg[dst, c] += h[src, c] * ex[e, c%8] / denom[dst, c%8]
// 32 lanes per edge, float4 per lane (coalesced gather; atomics hit L2).
// ---------------------------------------------------------------------------
__global__ __launch_bounds__(256)
void k_aggregate(const int* __restrict__ src, const int* __restrict__ dst,
                 const float* __restrict__ ex, const float* __restrict__ denom,
                 const float* __restrict__ h, float* __restrict__ agg)
{
    long idx = (long)blockIdx.x * 256 + threadIdx.x;
    if (idx >= (long)N_EDGES * 32) return;
    int e  = (int)(idx >> 5);
    int g  = (int)(idx & 31);
    int c0 = g * 4;            // column group; heads (c0&7)..(c0&7)+3
    int hb = c0 & 7;
    int s = src[e], d = dst[e];

    float4 hv = *(const float4*)&h[(size_t)s * DIM + c0];
    float4 e4 = *(const float4*)&ex[(size_t)e * NH + hb];
    const float* dn = &denom[(size_t)d * NH + hb];
    float p0 = e4.x / dn[0];
    float p1 = e4.y / dn[1];
    float p2 = e4.z / dn[2];
    float p3 = e4.w / dn[3];

    float* ag = &agg[(size_t)d * DIM + c0];
    atomicAdd(ag + 0, hv.x * p0);
    atomicAdd(ag + 1, hv.y * p1);
    atomicAdd(ag + 2, hv.z * p2);
    atomicAdd(ag + 3, hv.w * p3);
}

// ---------------------------------------------------------------------------
// Kernel 5: out = gelu(agg @ w1 + b1) @ w2 + b2   (two chained WMMA GEMMs)
// 64 rows/block. s_buf overlays: [f16 A | f16 T] during GEMMs, f32 OUT after.
// ---------------------------------------------------------------------------
__global__ __launch_bounds__(256)
void k_ffn(const float* __restrict__ agg, const float* __restrict__ b1,
           const float* __restrict__ b2,
           const _Float16* __restrict__ w1_t, const _Float16* __restrict__ w2_t,
           float* __restrict__ out)
{
    __shared__ __align__(32) char     s_buf[ROWS * DIM * 4];   // 32 KB overlay
    __shared__ __align__(32) _Float16 s_b1t[DIM * DIM];        // 32 KB
    __shared__ __align__(32) _Float16 s_b2t[DIM * DIM];        // 32 KB
    _Float16* s_a   = (_Float16*)s_buf;                        // [64][128] f16
    _Float16* s_t   = (_Float16*)(s_buf + ROWS * DIM * 2);     // [64][128] f16
    float*    s_out = (float*)s_buf;                           // [64][128] f32

    const int tid  = threadIdx.x;
    const int row0 = blockIdx.x * ROWS;

    for (int i = tid; i < ROWS * DIM; i += 256) {
        int r = i >> 7;
        s_a[i] = (row0 + r < N_NODES)
                   ? (_Float16)agg[(size_t)(row0 + r) * DIM + (i & 127)]
                   : (_Float16)0.0f;
    }
    {
        const uint4* w1s = (const uint4*)w1_t;  uint4* w1d = (uint4*)s_b1t;
        const uint4* w2s = (const uint4*)w2_t;  uint4* w2d = (uint4*)s_b2t;
        for (int i = tid; i < (DIM * DIM) / 8; i += 256) {
            w1d[i] = w1s[i];
            w2d[i] = w2s[i];
        }
    }
    __syncthreads();

    const int wave = tid >> 5, lane = tid & 31;
    const int n = lane & 15, mbase = (lane >> 4) * 8;
    const int col = wave * 16 + n;

    // layer 1 + tanh-approx GELU (jax.nn.gelu default) -> s_t (f16)
    v8f c1[4] = {};
    wmma_strip_64x16(s_a, s_b1t, wave, lane, c1);
    const float bias1 = b1[col];
#pragma unroll
    for (int rt = 0; rt < 4; ++rt)
#pragma unroll
        for (int i = 0; i < 8; ++i) {
            float v = c1[rt][i] + bias1;
            float g = 0.5f * v *
                (1.0f + tanhf(0.7978845608028654f * (v + 0.044715f * v * v * v)));
            s_t[(rt * 16 + mbase + i) * DIM + col] = (_Float16)g;
        }
    __syncthreads();

    // layer 2
    v8f c2[4] = {};
    wmma_strip_64x16(s_t, s_b2t, wave, lane, c2);
    __syncthreads();                       // all reads of s_a/s_t done

    const float bias2 = b2[col];
#pragma unroll
    for (int rt = 0; rt < 4; ++rt)
#pragma unroll
        for (int i = 0; i < 8; ++i)
            s_out[(rt * 16 + mbase + i) * DIM + col] = c2[rt][i] + bias2;
    __syncthreads();

    // cooperative coalesced store (float4 per lane)
    for (int i = tid; i < ROWS * (DIM / 4); i += 256) {
        int r = i >> 5, c4 = (i & 31) * 4;
        if (row0 + r < N_NODES)
            *(float4*)&out[(size_t)(row0 + r) * DIM + c4] =
                *(const float4*)&s_out[r * DIM + c4];
    }
}

// ---------------------------------------------------------------------------
// Host launcher
// ---------------------------------------------------------------------------
extern "C" void kernel_launch(void* const* d_in, const int* in_sizes, int n_in,
                              void* d_out, int out_size, void* d_ws, size_t ws_size,
                              hipStream_t stream)
{
    (void)in_sizes; (void)n_in; (void)out_size; (void)ws_size;
    const float* x    = (const float*)d_in[0];
    const int*   src  = (const int*)d_in[1];
    const int*   dst  = (const int*)d_in[2];
    const float* w_in = (const float*)d_in[3];
    const float* b_in = (const float*)d_in[4];
    const float* w_au = (const float*)d_in[5];
    const float* b_au = (const float*)d_in[6];
    const float* w_av = (const float*)d_in[7];
    const float* w1   = (const float*)d_in[8];
    const float* b1   = (const float*)d_in[9];
    const float* w2   = (const float*)d_in[10];
    const float* b2   = (const float*)d_in[11];
    float* out = (float*)d_out;

    char* p = (char*)d_ws;
    auto alloc = [&](size_t bytes) -> char* {
        char* r = p;
        p += (bytes + 255) & ~(size_t)255;
        return r;
    };
    float*    h      = (float*)   alloc((size_t)N_NODES * DIM * 4);
    float*    a_u    = (float*)   alloc((size_t)N_NODES * NH * 4);
    float*    a_v    = (float*)   alloc((size_t)N_NODES * NH * 4);
    unsigned* smx    = (unsigned*)alloc((size_t)N_NODES * NH * 4);
    float*    denom  = (float*)   alloc((size_t)N_NODES * NH * 4);
    float*    ex     = (float*)   alloc((size_t)N_EDGES * NH * 4);
    float*    agg    = (float*)   alloc((size_t)N_NODES * DIM * 4);
    _Float16* w_in_t = (_Float16*)alloc((size_t)DIM * DIM * 2);
    _Float16* w1_t   = (_Float16*)alloc((size_t)DIM * DIM * 2);
    _Float16* w2_t   = (_Float16*)alloc((size_t)DIM * DIM * 2);

    const int init_blocks = (N_NODES * DIM + 255) / 256;              // 50000
    k_init<<<init_blocks, 256, 0, stream>>>(w_in, w1, w2, w_in_t, w1_t, w2_t,
                                            smx, denom, agg);
    const int gemm_blocks = (N_NODES + ROWS - 1) / ROWS;              // 1563
    k_node_linear<<<gemm_blocks, 256, 0, stream>>>(x, b_in, w_au, b_au, w_av,
                                                   w_in_t, h, a_u, a_v);
    const int eh_blocks = (N_EDGES * NH + 255) / 256;                 // 50000
    k_edge_max<<<eh_blocks, 256, 0, stream>>>(src, dst, a_u, a_v, smx);
    k_edge_exp<<<eh_blocks, 256, 0, stream>>>(src, dst, a_u, a_v, smx, ex, denom);
    const int ag_blocks = (int)(((long)N_EDGES * 32 + 255) / 256);    // 200000
    k_aggregate<<<ag_blocks, 256, 0, stream>>>(src, dst, ex, denom, h, agg);
    k_ffn<<<gemm_blocks, 256, 0, stream>>>(agg, b1, b2, w1_t, w2_t, out);
}
